// TensorBoard_88313117540416
// MI455X (gfx1250) — matrix operand order repaired and checked by reading, not compiled, not run on hardware
//
#include <hip/hip_runtime.h>

// ---------------------------------------------------------------------------
// Go-board feature kernel for MI455X (gfx1250, wave32).
//
// One wave32 per 19x19 board; rows live as 19-bit masks in registers
// (lane r owns row r); cross-row traffic = ds_bpermute shuffles, no LDS for
// the board itself. Neighbor counts:  NB = T*E + E*T  (T = 32x32 tridiagonal,
// E = empty mask zero-padded) via 8x v_wmma_f32_16x16x32_f16 per board —
// exact for 0/1 inputs. Flood fill = bitwise dilation with uniform early
// exit, capped at 90 (monotone => bit-exact vs reference's fixed 90 iters).
// Super-ko history is copied LDS-ward with global_load_async_to_lds_b64 at
// kernel start and consumed after s_wait_asynccnt at the very end.
// ---------------------------------------------------------------------------

typedef __attribute__((ext_vector_type(16))) _Float16 v16h;
typedef __attribute__((ext_vector_type(8)))  float    v8f;
typedef __attribute__((ext_vector_type(8)))  unsigned v8u;

namespace {
constexpr int HH       = 19;
constexpr int CELLS    = HH * HH;         // 361
constexpr unsigned M19 = 0x7FFFFu;        // 19 valid bits per row
constexpr int MAXHIST  = 100;
constexpr int MAXFLOOD = 90;              // min(19*19/4, 100)
constexpr int WAVES    = 8;               // boards per 256-thread block
constexpr unsigned H1LO = 0x3C00u;        // f16 1.0 in low half-word
constexpr unsigned H1HI = 0x3C000000u;    // f16 1.0 in high half-word
}

__device__ __forceinline__ unsigned shflu(unsigned v, int src) {
  return (unsigned)__shfl((int)v, src, 32);
}

__device__ __forceinline__ long long shflxor64(long long v, int m) {
  int lo = (int)(v & 0xffffffffll);
  int hi = (int)(((unsigned long long)v) >> 32);
  lo = __shfl_xor(lo, m, 32);
  hi = __shfl_xor(hi, m, 32);
  return (long long)(((unsigned long long)(unsigned)hi << 32) | (unsigned)lo);
}

__global__ __launch_bounds__(256) void go_features_kernel(
    const float* __restrict__ board,        // [nB,19,19]
    const long long* __restrict__ zstones,  // [2,361]
    const long long* __restrict__ zturn,    // [2]
    const int* __restrict__ cur_player,     // [nB]
    const long long* __restrict__ history,  // [nB,100]
    float* __restrict__ outF,               // [nB,4,19,19]
    long long* __restrict__ outH,           // [nB]
    float* __restrict__ outK,               // [nB] 0/1
    int nB) {
  __shared__ long long hbuf[WAVES][MAXHIST];

  const int lane = threadIdx.x & 31;
  const int wv = threadIdx.x >> 5;
  const int b = blockIdx.x * WAVES + wv;
  if (b >= nB) return;  // uniform per wave -> EXEC stays all-1s for WMMA

  const long long* __restrict__ hist = history + (size_t)b * MAXHIST;

#if defined(__HIP_DEVICE_COMPILE__)
  // Kick off async copy of this board's history row into LDS; the transfer
  // overlaps the entire WMMA/flood/hash body below (tracked by ASYNCcnt).
  for (int i = lane; i < MAXHIST; i += 32) {
    unsigned laddr = (unsigned)(uintptr_t)(&hbuf[wv][i]);
    unsigned long long gaddr = (unsigned long long)(uintptr_t)(hist + i);
    asm volatile("global_load_async_to_lds_b64 %0, %1, off"
                 :: "v"(laddr), "v"(gaddr) : "memory");
  }
#endif
  __builtin_prefetch(zstones, 0, 2);  // global_prefetch_b8: warm shared table

  // ---- build bitmask rows: lane r owns row r (lanes 19..31 hold zeros) ----
  unsigned bm = 0, wm = 0, em = 0;
  if (lane < HH) {
    const float* rp = board + (size_t)b * CELLS + lane * HH;
    #pragma unroll
    for (int c = 0; c < HH; ++c) {
      float x = rp[c];
      bm |= (x > 0.65f ? 1u : 0u) << c;
      wm |= (x < 0.35f ? 1u : 0u) << c;
    }
    em = ~(bm | wm) & M19;   // padded rows (>=19) must stay 0
  }

  const int N = lane & 15;   // N-in-tile (B/D) == M-in-tile (A)
  const int g = lane >> 4;   // lane half

  // ---- B fragments (32x16, depend only on column tile tj) -----------------
  // B layout: VGPR v holds K = 16*g + 2v (lo16), +1 (hi16), at N = lane%16.
  unsigned m0s[8], m1s[8];
  #pragma unroll
  for (int v = 0; v < 8; ++v) {
    const int k0 = 16 * g + 2 * v;
    m0s[v] = shflu(em, k0);       // row k0 of E
    m1s[v] = shflu(em, k0 + 1);   // row k0+1 of E
  }
  v16h bEf[2], bTf[2];
  #pragma unroll
  for (int tj = 0; tj < 2; ++tj) {
    const int c = tj * 16 + N;
    v8u be, bt;
    #pragma unroll
    for (int v = 0; v < 8; ++v) {
      const int k0 = 16 * g + 2 * v;
      // E[k0][c] (lo) | E[k0+1][c] (hi), as f16 1.0/0.0 via integer multiply
      be[v] = (((m0s[v] >> c) & 1u) | (((m1s[v] >> c) & 1u) << 16)) * H1LO;
      // T[k][c]: lo/hi never both set; select on parity of (k0-c+2) in [0,4)
      const unsigned du = (unsigned)(k0 - c + 2);
      bt[v] = (du < 4u) ? ((du & 1u) ? H1LO : H1HI) : 0u;
    }
    bEf[tj] = __builtin_bit_cast(v16h, be);
    bTf[tj] = __builtin_bit_cast(v16h, bt);
  }

  // ---- A fragments per row tile + 8x WMMA:  NB = T*E + E*T ---------------
  // A layout: lane -> M = lane%16, K-half = lane/16;
//           VGPR v holds K = 16*half + 2*v (lo16), +1 (hi16).
  v8f dacc[2][2];
  #pragma unroll
  for (int ti = 0; ti < 2; ++ti) {
    const int r = ti * 16 + N;
    const unsigned emr = shflu(em, r);  // row r of E
    v8u ae, at;
    #pragma unroll
    for (int v = 0; v < 8; ++v) {
      const int kb = 16 * g + 2 * v;
      ae[v] = (((emr >> kb) & 1u) | (((emr >> (kb + 1)) & 1u) << 16)) * H1LO;
      const unsigned du = (unsigned)(r - kb + 1);
      at[v] = (du < 4u) ? ((du & 1u) ? H1HI : H1LO) : 0u;
    }
    const v16h aEf = __builtin_bit_cast(v16h, ae);
    const v16h aTf = __builtin_bit_cast(v16h, at);
    #pragma unroll
    for (int tj = 0; tj < 2; ++tj) {
      v8f d = {};
      d = __builtin_amdgcn_wmma_f32_16x16x32_f16(false, aTf, false, bEf[tj],
                                                 (short)0, d, false, false);
      d = __builtin_amdgcn_wmma_f32_16x16x32_f16(false, aEf, false, bTf[tj],
                                                 (short)0, d, false, false);
      dacc[ti][tj] = d;
    }
  }

  // ---- flood fill: seed = stones & dilate4(empty); iterate <= 90 ----------
  const unsigned de = (((em << 1) | (em >> 1)) & M19)
                    | shflu(em, (lane + 31) & 31)   // above (lane0<-lane31==0)
                    | shflu(em, (lane + 1) & 31);   // below (lane18<-lane19==0)
  unsigned ab = bm & de, aw = wm & de;
  for (int it = 0; it < MAXFLOOD; ++it) {
    const unsigned ub = shflu(ab, (lane + 31) & 31), db = shflu(ab, (lane + 1) & 31);
    const unsigned uw = shflu(aw, (lane + 31) & 31), dw = shflu(aw, (lane + 1) & 31);
    const unsigned nab = ab | (((((ab << 1) | (ab >> 1)) & M19) | ub | db) & bm);
    const unsigned naw = aw | (((((aw << 1) | (aw >> 1)) & M19) | uw | dw) & wm);
    const int ch = (nab != ab) | (naw != aw);
    ab = nab; aw = naw;
    if (!__any(ch)) break;   // fixed point: identical to running all 90 iters
  }

  // ---- Zobrist hash: per-row XOR, butterfly xor-reduce across the wave ----
  long long h = 0;
  if (lane < HH) {
    const int base = lane * HH;
    #pragma unroll
    for (int c = 0; c < HH; ++c) {
      if ((bm >> c) & 1u) h ^= zstones[base + c];
      if ((wm >> c) & 1u) h ^= zstones[CELLS + base + c];
    }
  }
  #pragma unroll
  for (int off = 16; off; off >>= 1) h ^= shflxor64(h, off);
  h ^= zturn[cur_player[b]];               // uniform; every lane holds hash

  // ---- super-ko scan from LDS (async copy finished long ago) --------------
  int mk = 0;
#if defined(__HIP_DEVICE_COMPILE__)
  asm volatile("s_wait_asynccnt 0x0" ::: "memory");
  for (int i = lane; i < MAXHIST; i += 32) mk |= (hbuf[wv][i] == h);
#else
  for (int i = lane; i < MAXHIST; i += 32) mk |= (hist[i] == h);
#endif
  const int sk = __any(mk);

  // ---- stores: D-fragment geometry -> 16 consecutive columns per half -----
  float* __restrict__ ob = outF + (size_t)b * 4 * CELLS;
  #pragma unroll
  for (int ti = 0; ti < 2; ++ti) {
    #pragma unroll
    for (int v = 0; v < 8; ++v) {
      const int rr = ti * 16 + v + 8 * g;
      // shuffles before divergence (disabled-lane reads return 0)
      const unsigned bmr = shflu(bm, rr & 31);
      const unsigned wmr = shflu(wm, rr & 31);
      const unsigned abr = shflu(ab, rr & 31);
      const unsigned awr = shflu(aw, rr & 31);
      #pragma unroll
      for (int tj = 0; tj < 2; ++tj) {
        const int cc = tj * 16 + N;
        if (rr < HH && cc < HH) {
          const int o = rr * HH + cc;
          const float nb = dacc[ti][tj][v];                    // nb_empty
          ob[o]             = ((bmr >> cc) & 1u) ? nb : 0.0f;  // lib black
          ob[o + CELLS]     = ((wmr >> cc) & 1u) ? nb : 0.0f;  // lib white
          ob[o + 2 * CELLS] = (float)((abr >> cc) & 1u);       // alive black
          ob[o + 3 * CELLS] = (float)((awr >> cc) & 1u);       // alive white
        }
      }
    }
  }

  if (lane == 0) {
    outH[b] = h;
    outK[b] = sk ? 1.0f : 0.0f;
  }
}

extern "C" void kernel_launch(void* const* d_in, const int* in_sizes, int n_in,
                              void* d_out, int out_size, void* d_ws, size_t ws_size,
                              hipStream_t stream) {
  const float*     board = (const float*)d_in[0];
  const long long* zs    = (const long long*)d_in[1];
  const long long* zt    = (const long long*)d_in[2];
  const int*       cp    = (const int*)d_in[3];
  const long long* ph    = (const long long*)d_in[4];

  const int nB = in_sizes[0] / CELLS;   // 32768

  // d_out layout (tuple concatenated in return order):
  //   [nB*4*361] f32 lib/alive, then [nB] int64 hashes, then [nB] f32 superko
  float*     outF = (float*)d_out;
  long long* outH = (long long*)(outF + (size_t)nB * 4 * CELLS);
  float*     outK = (float*)(outH + nB);

  dim3 block(32 * WAVES);               // 8 waves = 8 boards per block
  dim3 grid((nB + WAVES - 1) / WAVES);
  go_features_kernel<<<grid, block, 0, stream>>>(board, zs, zt, cp, ph,
                                                 outF, outH, outK, nB);
}
